// SelfAttention_42700564857219
// MI455X (gfx1250) — matrix operand order repaired and checked by reading, not compiled, not run on hardware
//
#include <hip/hip_runtime.h>
#include <hip/hip_bf16.h>

typedef __attribute__((ext_vector_type(16))) __bf16 v16bf;
typedef __attribute__((ext_vector_type(8)))  __bf16 v8bf;
typedef __attribute__((ext_vector_type(8)))  float  v8f;

#define B_  32
#define C_  512
#define N_  1024
#define C8_ 64

// ---------------- WMMA helpers (wave32, gfx1250) ----------------

__device__ __forceinline__ v8f wmma_bf16(v16bf a, v16bf b, v8f c) {
  // D = A(16x32) * B(32x16) + C, f32 accumulate
  return __builtin_amdgcn_wmma_f32_16x16x32_bf16(false, a, false, b, (short)0, c, false, false);
}

// A-fragment: lane&15 = row M; kh = lane>>4 picks interleaved K chunks
// elements 0..7 -> K = kbase + kh*8 + i ; elements 8..15 -> K = kbase + 16 + kh*8 + i
__device__ __forceinline__ v16bf load_a_frag(const __bf16* row, int kbase, int lane) {
  const int kh = (lane >> 4) & 1;
  v8bf lo = *(const v8bf*)(row + kbase + kh * 8);
  v8bf hi = *(const v8bf*)(row + kbase + 16 + kh * 8);
  v16bf f;
#pragma unroll
  for (int i = 0; i < 8; ++i) { f[i] = lo[i]; f[i + 8] = hi[i]; }
  return f;
}

// B-fragment: lane&15 = output column; kh = lane>>4 picks interleaved K chunks
// elements 0..7 -> K = kbase + kh*8 + i ; elements 8..15 -> K = kbase + 16 + kh*8 + i
__device__ __forceinline__ v16bf load_b_frag(const __bf16* row, int kbase, int lane) {
  const int kh = (lane >> 4) & 1;
  v8bf lo = *(const v8bf*)(row + kbase + kh * 8);
  v8bf hi = *(const v8bf*)(row + kbase + 16 + kh * 8);
  v16bf f;
#pragma unroll
  for (int i = 0; i < 8; ++i) { f[i] = lo[i]; f[i + 8] = hi[i]; }
  return f;
}

// ---------------- small prep kernels ----------------

__global__ void k_convert_bf16(const float* __restrict__ in, __bf16* __restrict__ out, int n) {
  int i = blockIdx.x * blockDim.x + threadIdx.x;
  if (i < n) out[i] = (__bf16)in[i];
}

// x f32 [B, C, N] -> Xt bf16 [B, N, C]
__global__ void k_transpose_x(const float* __restrict__ x, __bf16* __restrict__ xt) {
  int i = blockIdx.x * blockDim.x + threadIdx.x;  // over B*C*N, coalesced read
  int n = i & (N_ - 1);
  int c = (i >> 10) & (C_ - 1);
  int b = i >> 19;
  xt[((size_t)b * N_ + n) * C_ + c] = (__bf16)x[i];
}

// ---------------- projections ----------------

// Q/K: Out[b,n,o] = Xt_row(n) . W_row(o) + bias[o]   (O = 64)
__global__ void k_proj_qk(const __bf16* __restrict__ Xt, const __bf16* __restrict__ W,
                          const float* __restrict__ bias, __bf16* __restrict__ Out) {
  const int lane = threadIdx.x & 31;
  const int wt = blockIdx.x * (blockDim.x >> 5) + (threadIdx.x >> 5);
  const int ot = wt & 3;           // 64/16
  const int nt = (wt >> 2) & 63;   // 1024/16
  const int b  = wt >> 8;
  const __bf16* arow = Xt + ((size_t)b * N_ + nt * 16 + (lane & 15)) * C_;
  const __bf16* brow = W + (size_t)(ot * 16 + (lane & 15)) * C_;
  v8f acc = {};
#pragma unroll 4
  for (int kb = 0; kb < C_; kb += 32)
    acc = wmma_bf16(load_a_frag(arow, kb, lane), load_b_frag(brow, kb, lane), acc);
  const float bv = bias[ot * 16 + (lane & 15)];
  const int m0 = (lane >> 4) * 8;
#pragma unroll
  for (int r = 0; r < 8; ++r)
    Out[((size_t)b * N_ + nt * 16 + m0 + r) * C8_ + ot * 16 + (lane & 15)] = (__bf16)(acc[r] + bv);
}

// V: V[b,c,n] = Wv_row(c) . Xt_row(n) + bv[c]
__global__ void k_proj_v(const __bf16* __restrict__ Xt, const __bf16* __restrict__ Wv,
                         const float* __restrict__ bv, __bf16* __restrict__ V) {
  const int lane = threadIdx.x & 31;
  const int wt = blockIdx.x * (blockDim.x >> 5) + (threadIdx.x >> 5);
  const int nt = wt & 63;          // 1024/16
  const int ct = (wt >> 6) & 31;   // 512/16
  const int b  = wt >> 11;
  const __bf16* arow = Wv + (size_t)(ct * 16 + (lane & 15)) * C_;
  const __bf16* brow = Xt + ((size_t)b * N_ + nt * 16 + (lane & 15)) * C_;
  v8f acc = {};
#pragma unroll 4
  for (int kb = 0; kb < C_; kb += 32)
    acc = wmma_bf16(load_a_frag(arow, kb, lane), load_b_frag(brow, kb, lane), acc);
  const int m0 = (lane >> 4) * 8;
#pragma unroll
  for (int r = 0; r < 8; ++r) {
    int c = ct * 16 + m0 + r;
    V[((size_t)b * C_ + c) * N_ + nt * 16 + (lane & 15)] = (__bf16)(acc[r] + bv[c]);
  }
}

// ---------------- pass 1: softmax row stats (online max / sum) ----------------

__global__ void k_stats(const __bf16* __restrict__ Q, const __bf16* __restrict__ K,
                        float2* __restrict__ stats) {
  const int lane = threadIdx.x & 31;
  const int wt = blockIdx.x * (blockDim.x >> 5) + (threadIdx.x >> 5);
  const int qt = wt & 63;
  const int b  = wt >> 6;
  const __bf16* qrow = Q + ((size_t)b * N_ + qt * 16 + (lane & 15)) * C8_;
  v16bf qa0 = load_a_frag(qrow, 0, lane);
  v16bf qa1 = load_a_frag(qrow, 32, lane);
  float M[8], L[8];
#pragma unroll
  for (int r = 0; r < 8; ++r) { M[r] = -3.0e38f; L[r] = 0.0f; }
  const __bf16* Kb = K + (size_t)b * N_ * C8_;
  for (int kt = 0; kt < 64; ++kt) {
    const __bf16* krow = Kb + (size_t)(kt * 16 + (lane & 15)) * C8_;
    v8f s = {};
    s = wmma_bf16(qa0, load_b_frag(krow, 0, lane), s);
    s = wmma_bf16(qa1, load_b_frag(krow, 32, lane), s);
#pragma unroll
    for (int r = 0; r < 8; ++r) {
      float t = s[r];
#pragma unroll
      for (int msk = 1; msk < 16; msk <<= 1) t = fmaxf(t, __shfl_xor(t, msk, 32));
      float mn = fmaxf(M[r], t);
      float e = __expf(s[r] - mn);
#pragma unroll
      for (int msk = 1; msk < 16; msk <<= 1) e += __shfl_xor(e, msk, 32);
      L[r] = L[r] * __expf(M[r] - mn) + e;
      M[r] = mn;
    }
  }
  if ((lane & 15) == 0) {
    const int m0 = (lane >> 4) * 8;
#pragma unroll
    for (int r = 0; r < 8; ++r) {
      float2 st; st.x = M[r]; st.y = 1.0f / L[r];
      stats[(size_t)b * N_ + qt * 16 + m0 + r] = st;
    }
  }
}

// ---------------- pass 2: cooperative softmax-recompute + PV + coalesced epilogue ----
// Block = 4 waves, all on one 16-query tile. Wave w computes p for key chunk
// mb+32w..+31 (energy computed once per block), stages via LDS; every wave then
// consumes all 4 p-fragments for its own 128 output channels.

#define CW 4
__global__ void k_attn_out(const __bf16* __restrict__ Q, const __bf16* __restrict__ K,
                           const __bf16* __restrict__ V, const float2* __restrict__ stats,
                           const float* __restrict__ x, const float* __restrict__ alphap,
                           float* __restrict__ out) {
  __shared__ __align__(32) __bf16 pstage[CW][16 * 32];   // 4 KB: p tiles (shared across waves)
  __shared__ __align__(16) float  fstage[CW][8 * 256];   // 32 KB: per-wave epilogue transpose
  const int lane = threadIdx.x & 31;
  const int wid  = threadIdx.x >> 5;  // channel group AND key-chunk owner
  const int qt = blockIdx.x & 63;
  const int b  = blockIdx.x >> 6;
  const int m0 = (lane >> 4) * 8;

  const __bf16* qrow = Q + ((size_t)b * N_ + qt * 16 + (lane & 15)) * C8_;
  v16bf qa0 = load_a_frag(qrow, 0, lane);
  v16bf qa1 = load_a_frag(qrow, 32, lane);

  float Mr[8], Ir[8];
#pragma unroll
  for (int r = 0; r < 8; ++r) {
    float2 st = stats[(size_t)b * N_ + qt * 16 + m0 + r];
    Mr[r] = st.x; Ir[r] = st.y;
  }

  const __bf16* Kb = K + (size_t)b * N_ * C8_;
  const __bf16* Vb = V + (size_t)b * C_ * N_;

  v8f acc[8];
#pragma unroll
  for (int ct = 0; ct < 8; ++ct)
#pragma unroll
    for (int r = 0; r < 8; ++r) acc[ct][r] = 0.0f;

  for (int mb = 0; mb < N_; mb += 32 * CW) {
    // --- this wave's 16x32 probability tile for keys [mb+32*wid, +32) ---
#pragma unroll
    for (int st2 = 0; st2 < 2; ++st2) {
      const __bf16* krow = Kb + (size_t)(mb + wid * 32 + st2 * 16 + (lane & 15)) * C8_;
      v8f s = {};
      s = wmma_bf16(qa0, load_b_frag(krow, 0, lane), s);
      s = wmma_bf16(qa1, load_b_frag(krow, 32, lane), s);
#pragma unroll
      for (int r = 0; r < 8; ++r) {
        float p = __expf(s[r] - Mr[r]) * Ir[r];
        pstage[wid][(m0 + r) * 32 + st2 * 16 + (lane & 15)] = (__bf16)p;  // D-layout -> [q][k]
      }
    }
    if (mb + 32 * CW < N_)  // prefetch next key chunk (global_prefetch_b8)
      __builtin_prefetch(Kb + (size_t)(mb + 32 * CW + wid * 32 + (lane & 15)) * C8_, 0, 3);
    __syncthreads();
    // --- PV: consume all 4 p tiles for this wave's 128 channels ---
#pragma unroll
    for (int j = 0; j < CW; ++j) {
      v16bf pf = load_a_frag(&pstage[j][(lane & 15) * 32], 0, lane);
#pragma unroll
      for (int ct = 0; ct < 8; ++ct) {
        const __bf16* vrow = Vb + (size_t)(wid * 128 + ct * 16 + (lane & 15)) * N_;
        acc[ct] = wmma_bf16(pf, load_b_frag(vrow, mb + j * 32, lane), acc[ct]);
      }
    }
    __syncthreads();
  }

  // --- epilogue: transpose D tiles through wave-private LDS, float4-coalesced I/O ---
  const float alpha = *alphap;
  float* fw = &fstage[wid][0];
#pragma unroll
  for (int ct = 0; ct < 8; ++ct)
#pragma unroll
    for (int r = 0; r < 8; ++r)
      fw[ct * 256 + (lane & 15) * 16 + m0 + r] = acc[ct][r];  // [c_local][q]
  asm volatile("" ::: "memory");  // keep DS loads after DS stores (HW is in-order per wave)
#pragma unroll
  for (int ct = 0; ct < 8; ++ct) {
#pragma unroll
    for (int it = 0; it < 2; ++it) {
      const int cl = (lane >> 2) + it * 8;   // local channel row
      const int qp = lane & 3;               // 4 floats per lane
      const float4 t = *(const float4*)&fw[ct * 256 + cl * 16 + qp * 4];
      const size_t gidx = ((size_t)b * C_ + wid * 128 + ct * 16 + cl) * N_ + qt * 16 + qp * 4;
      const float4 xv = *(const float4*)&x[gidx];
      float4 o;
      o.x = alpha * t.x + xv.x;
      o.y = alpha * t.y + xv.y;
      o.z = alpha * t.z + xv.z;
      o.w = alpha * t.w + xv.w;
      *(float4*)&out[gidx] = o;
    }
  }
}

// ---------------- launch ----------------

extern "C" void kernel_launch(void* const* d_in, const int* in_sizes, int n_in,
                              void* d_out, int out_size, void* d_ws, size_t ws_size,
                              hipStream_t stream) {
  const float* x     = (const float*)d_in[0];
  const float* Wq    = (const float*)d_in[1];
  const float* bq    = (const float*)d_in[2];
  const float* Wk    = (const float*)d_in[3];
  const float* bk    = (const float*)d_in[4];
  const float* Wv    = (const float*)d_in[5];
  const float* bv    = (const float*)d_in[6];
  const float* alpha = (const float*)d_in[7];
  float* out = (float*)d_out;

  char* ws = (char*)d_ws;
  size_t off = 0;
  auto take = [&](size_t bytes) -> char* {
    char* p = ws + off;
    off += (bytes + 255) & ~(size_t)255;
    return p;
  };
  // total workspace ~76.5 MB
  __bf16* Xt  = (__bf16*)take((size_t)B_ * N_ * C_ * 2);   // x transposed, bf16 [B,N,C]
  __bf16* Wqb = (__bf16*)take((size_t)C8_ * C_ * 2);
  __bf16* Wkb = (__bf16*)take((size_t)C8_ * C_ * 2);
  __bf16* Wvb = (__bf16*)take((size_t)C_ * C_ * 2);
  __bf16* Qw  = (__bf16*)take((size_t)B_ * N_ * C8_ * 2);  // [B,N,64]
  __bf16* Kw  = (__bf16*)take((size_t)B_ * N_ * C8_ * 2);  // [B,N,64]
  __bf16* Vw  = (__bf16*)take((size_t)B_ * C_ * N_ * 2);   // [B,C,N]
  float2* St  = (float2*)take((size_t)B_ * N_ * sizeof(float2));
  (void)ws_size; (void)in_sizes; (void)n_in; (void)out_size;

  k_convert_bf16<<<(C8_ * C_ + 255) / 256, 256, 0, stream>>>(Wq, Wqb, C8_ * C_);
  k_convert_bf16<<<(C8_ * C_ + 255) / 256, 256, 0, stream>>>(Wk, Wkb, C8_ * C_);
  k_convert_bf16<<<(C_ * C_ + 255) / 256, 256, 0, stream>>>(Wv, Wvb, C_ * C_);
  k_transpose_x<<<(B_ * C_ * N_) / 256, 256, 0, stream>>>(x, Xt);

  k_proj_qk<<<(B_ * 64 * 4) / 4, 128, 0, stream>>>(Xt, Wqb, bq, Qw);
  k_proj_qk<<<(B_ * 64 * 4) / 4, 128, 0, stream>>>(Xt, Wkb, bk, Kw);
  k_proj_v<<<(B_ * 32 * 64) / 4, 128, 0, stream>>>(Xt, Wvb, bv, Vw);

  k_stats<<<(B_ * 64) / 4, 128, 0, stream>>>(Qw, Kw, St);
  k_attn_out<<<B_ * 64, 32 * CW, 0, stream>>>(Qw, Kw, Vw, St, x, alpha, out);
}
